// Rnn_88854283420042
// MI455X (gfx1250) — matrix operand order repaired and checked by reading, not compiled, not run on hardware
//
#include <hip/hip_runtime.h>
#include <stdint.h>

// GRU: T=2048, B=64, D=256, H=256, 3H=768
#define T_STEPS 2048
#define B_SZ    64
#define D_SZ    256
#define H_SZ    256
#define G3      768
#define TBROWS  (T_STEPS * B_SZ)   // 131072
#define SLAB    (B_SZ * G3)        // 49152 elements per timestep of xp

typedef __attribute__((ext_vector_type(16))) __bf16 v16bf;
typedef __attribute__((ext_vector_type(8)))  __bf16 v8bf;
typedef __attribute__((ext_vector_type(8)))  float  v8f;
typedef __attribute__((ext_vector_type(4)))  unsigned int v4u;
typedef __attribute__((ext_vector_type(8)))  int    v8i;
typedef __attribute__((ext_vector_type(4)))  int    v4i;

__device__ __forceinline__ v8f wmma_bf16(v16bf a, v16bf b, v8f c) {
    // D = A(16x32 bf16) x B(32x16 bf16) + C(16x16 f32)
    return __builtin_amdgcn_wmma_f32_16x16x32_bf16(
        false, a, false, b, (short)0, c, false, false);
}

__device__ __forceinline__ float sigmoid_f(float x) {
    return 1.0f / (1.0f + __expf(-x));
}

// ---------------------------------------------------------------------------
// TDM: issue an async DMA of one xp timestep slab (49152 bf16 = 96KB, linear)
// from global memory into LDS.  D# per CDNA5 ISA ch.8:
//   group0: count=1 | lds_addr[63:32] | global_addr[120:64] | type=2[127:126]
//   group1: data_size=1(2B), tensor_dim0=tile_dim0=49152, tensor/tile_dim1=1,
//           tensor_dim0_stride=49152, no pad / iterate / multicast
//   group2/3: zero (2D tensor)
// Tracked by TENSORcnt (s_wait_tensorcnt).  6-arg builtin form (clang-23 /
// therock-10.0 lane): (v4u, v8i, v4i, v4i, v8i, i32 cpol).
// ---------------------------------------------------------------------------
__device__ __forceinline__ void tdm_load_slab(const __bf16* gsrc, unsigned lds_off) {
    unsigned long long ga = (unsigned long long)(uintptr_t)gsrc;
    v4u g0;
    g0[0] = 1u;                                            // count=1, user mode
    g0[1] = lds_off;                                       // lds_addr (bytes)
    g0[2] = (unsigned)(ga & 0xffffffffu);                  // global_addr[31:0]
    g0[3] = (unsigned)((ga >> 32) & 0x1ffffffu) | (2u << 30); // [56:32] | type=2
    v8i g1;
    g1[0] = 0x00010000;          // workgroup_mask=0, data_size=1 (2 bytes)
    g1[1] = (int)0xC0000000u;    // tensor_dim0[15:0]=49152 in bits 63:48
    g1[2] = 0x00010000;          // tensor_dim0[31:16]=0, tensor_dim1=1
    g1[3] = (int)0xC0000000u;    // tile_dim0=49152 in bits 127:112
    g1[4] = 0x00000001;          // tile_dim1=1, tile_dim2=0
    g1[5] = 0x0000C000;          // tensor_dim0_stride=49152 (bits 191:160)
    g1[6] = 0;
    g1[7] = 0;
    v4i gz4 = {0, 0, 0, 0};
    v8i gz8 = {0, 0, 0, 0, 0, 0, 0, 0};
    __builtin_amdgcn_tensor_load_to_lds(g0, g1, gz4, gz4, gz8, 0);
}

// ---------------------------------------------------------------------------
// Pack a (K=256) x (N=768) row-major f32 weight into bf16 WMMA-B fragments.
// Fragment (nt, kt): nt in [0,48), kt in [0,8).
//   n = nt*16 + (lane & 15),  k = kt*32 + (lane<16 ? 0 : 8) + (e & 7) + (e >> 3)*16,  e in [0,16)
// dst element index = ((nt*8 + kt)*32 + lane)*16 + e   (one v16bf per lane)
// ---------------------------------------------------------------------------
__global__ void pack_w_kernel(const float* __restrict__ W, __bf16* __restrict__ dst) {
    int tid  = blockIdx.x * blockDim.x + threadIdx.x;   // 196608 total
    int e    = tid & 15;
    int lane = (tid >> 4) & 31;
    int frag = tid >> 9;
    int nt   = frag >> 3;
    int kt   = frag & 7;
    int n = nt * 16 + (lane & 15);
    int k = kt * 32 + ((lane < 16) ? 0 : 8) + (e & 7) + ((e >> 3) * 16);
    dst[tid] = (__bf16)W[k * G3 + n];
}

// ---------------------------------------------------------------------------
// Kernel A: xp = bf16(x @ Wx + bx)   (131072 x 256) * (256 x 768)
// One WG (256 thr = 8 waves) owns 64 rows x all 768 cols.
// A fragments loaded once per wave (global f32 -> bf16), reused for 24 tiles.
// ---------------------------------------------------------------------------
__global__ __launch_bounds__(256) void xp_gemm_kernel(
        const float* __restrict__ x, const __bf16* __restrict__ WxT,
        const float* __restrict__ bx, __bf16* __restrict__ xp) {
    const int lane  = threadIdx.x & 31;
    const int wave  = threadIdx.x >> 5;      // 0..7
    const int mtile = wave & 3;
    const int nhalf = wave >> 2;             // 0..1
    const int arow  = blockIdx.x * 64 + mtile * 16 + (lane & 15);
    const int koff  = (lane < 16) ? 0 : 8;
    const int coff  = (lane < 16) ? 0 : 8;
    const v16bf* __restrict__ Wb = (const v16bf*)WxT;

    v16bf afr[8];
#pragma unroll
    for (int kt = 0; kt < 8; ++kt) {
        const float* xr = x + arow * D_SZ + kt * 32 + koff;
        v16bf a;
#pragma unroll
        for (int e = 0; e < 8; ++e) {
            a[e]     = (__bf16)xr[e];
            a[e + 8] = (__bf16)xr[16 + e];
        }
        afr[kt] = a;
    }

    const int ntbase = nhalf * 24;
#pragma unroll 1
    for (int ni = 0; ni < 24; ++ni) {
        const int nt = ntbase + ni;
        v8f acc = {};
#pragma unroll
        for (int kt = 0; kt < 8; ++kt) {
            v16bf b = Wb[(nt * 8 + kt) * 32 + lane];
            acc = wmma_bf16(afr[kt], b, acc);
        }
        const int col = nt * 16 + (lane & 15);
        const float bxi = bx[col];
#pragma unroll
        for (int i = 0; i < 8; ++i) {
            const int r = blockIdx.x * 64 + mtile * 16 + i + coff;
            xp[(size_t)r * G3 + col] = (__bf16)(acc[i] + bxi);
        }
    }
}

// ---------------------------------------------------------------------------
// Kernel B: persistent single-WG GRU scan.
//  * 512 threads = 16 waves; wave w == jtile w.  Each wave: 4 mtiles x 3 gates
//    = 12 independent 8-deep WMMA chains; each Wh fragment read ONCE per step.
//  * h: bf16 ping-pong in LDS (2 x 32KB); old-h carried in registers.
//  * xp: streamed one step ahead by the Tensor Data Mover into an LDS
//    ping-pong slab (2 x 96KB).  Wave 0 issues tensor_load_to_lds for t+1,
//    waits TENSORcnt<=1 (step t resident, t+1 in flight), then a barrier
//    publishes the slab.  Total LDS = 256KB of the 320KB WGP budget.
// ---------------------------------------------------------------------------
__global__ __launch_bounds__(512, 1) void gru_scan_kernel(
        const __bf16* __restrict__ xp, const __bf16* __restrict__ WhT,
        const float* __restrict__ bh, const float* __restrict__ h0,
        float* __restrict__ out_carry, float* __restrict__ out_seq) {
    __shared__ __bf16 hb[2][B_SZ * H_SZ];    // 2 x 32KB
    __shared__ __bf16 xbuf[2][SLAB];         // 2 x 96KB

    const int tid  = threadIdx.x;
    const int lane = tid & 31;
    const int wave = tid >> 5;               // == jtile, 0..15
    const int l15  = lane & 15;
    const int hihalf = lane >> 4;
    const int koff = hihalf ? 8 : 0;
    const int coff = hihalf ? 8 : 0;
    const int col  = wave * 16 + l15;

    // prime the TDM pipeline with slab t=0
    if (wave == 0)
        tdm_load_slab(xp, (unsigned)(uintptr_t)&xbuf[0][0]);

    for (int i = tid; i < B_SZ * H_SZ; i += 512)
        hb[0][i] = (__bf16)h0[i];

    float hreg[4][8];
#pragma unroll
    for (int m = 0; m < 4; ++m)
#pragma unroll
        for (int i = 0; i < 8; ++i)
            hreg[m][i] = h0[(m * 16 + i + coff) * H_SZ + col];

    const float bhr = bh[col];
    const float bhz = bh[H_SZ + col];
    const float bhn = bh[2 * H_SZ + col];
    const v16bf* __restrict__ Wb = (const v16bf*)WhT;

    __syncthreads();

#pragma unroll 1
    for (int t = 0; t < T_STEPS; ++t) {
        // (a) kick off async DMA of next step's xp slab (TENSORcnt++)
        if (wave == 0 && t + 1 < T_STEPS)
            tdm_load_slab(xp + (size_t)(t + 1) * SLAB,
                          (unsigned)(uintptr_t)&xbuf[(t + 1) & 1][0]);

        // (b) recurrent GEMM: hp = h @ Wh  (bf16 WMMA, f32 accum)
        const __bf16* __restrict__ hcur = &hb[t & 1][0];
        v8f acc[4][3] = {};
#pragma unroll
        for (int kt = 0; kt < 8; ++kt) {
            v16bf br = Wb[(((0 * 16 + wave) * 8) + kt) * 32 + lane];
            v16bf bz = Wb[(((1 * 16 + wave) * 8) + kt) * 32 + lane];
            v16bf bn = Wb[(((2 * 16 + wave) * 8) + kt) * 32 + lane];
#pragma unroll
            for (int m = 0; m < 4; ++m) {
                const __bf16* hr = &hcur[(m * 16 + l15) * H_SZ + kt * 32 + koff];
                v8bf lo = *(const v8bf*)hr;
                v8bf hi = *(const v8bf*)(hr + 16);
                v16bf a = __builtin_shufflevector(lo, hi,
                        0, 1, 2, 3, 4, 5, 6, 7, 8, 9, 10, 11, 12, 13, 14, 15);
                acc[m][0] = wmma_bf16(a, br, acc[m][0]);
                acc[m][1] = wmma_bf16(a, bz, acc[m][1]);
                acc[m][2] = wmma_bf16(a, bn, acc[m][2]);
            }
        }

        // (c) step-t slab must be resident (t+1 may still be in flight)
        if (wave == 0)
            __builtin_amdgcn_s_wait_tensorcnt(1);
        __syncthreads();   // publish slab t; also orders h reads vs h writes

        // (d) gate elementwise + state update
        const __bf16* __restrict__ xpt = &xbuf[t & 1][0];
        __bf16* __restrict__ hnxt = &hb[(t + 1) & 1][0];
#pragma unroll
        for (int m = 0; m < 4; ++m) {
#pragma unroll
            for (int i = 0; i < 8; ++i) {
                const int row = m * 16 + i + coff;
                const float xr = (float)xpt[row * G3 + col];
                const float xz = (float)xpt[row * G3 + H_SZ + col];
                const float xn = (float)xpt[row * G3 + 2 * H_SZ + col];
                const float r = sigmoid_f(xr + acc[m][0][i] + bhr);
                const float z = sigmoid_f(xz + acc[m][1][i] + bhz);
                const float n = tanhf(xn + r * (acc[m][2][i] + bhn));
                const float hnew = (1.0f - z) * n + z * hreg[m][i];
                hreg[m][i] = hnew;
                hnxt[row * H_SZ + col] = (__bf16)hnew;
                out_seq[(size_t)t * (B_SZ * H_SZ) + row * H_SZ + col] = hnew;
            }
        }
        __syncthreads();   // h(t+1) complete; slab buffer (t&1) free for reuse
    }

#pragma unroll
    for (int m = 0; m < 4; ++m)
#pragma unroll
        for (int i = 0; i < 8; ++i)
            out_carry[(m * 16 + i + coff) * H_SZ + col] = hreg[m][i];
}

// ---------------------------------------------------------------------------
extern "C" void kernel_launch(void* const* d_in, const int* in_sizes, int n_in,
                              void* d_out, int out_size, void* d_ws, size_t ws_size,
                              hipStream_t stream) {
    const float* x  = (const float*)d_in[0];   // (T,B,D)
    const float* Wx = (const float*)d_in[1];   // (D,768)
    const float* Wh = (const float*)d_in[2];   // (H,768)
    const float* bx = (const float*)d_in[3];   // (768)
    const float* bh = (const float*)d_in[4];   // (768)
    const float* h0 = (const float*)d_in[5];   // (B,H)
    float* out = (float*)d_out;                // [carry (B*H)] ++ [seq (T*B*H)]

    char* ws = (char*)d_ws;
    __bf16* xp  = (__bf16*)ws;                                 // 131072*768*2 B
    __bf16* WxT = (__bf16*)(ws + (size_t)TBROWS * G3 * sizeof(__bf16));
    __bf16* WhT = WxT + (size_t)G3 * D_SZ;                     // 196608 elems each

    pack_w_kernel<<<768, 256, 0, stream>>>(Wx, WxT);
    pack_w_kernel<<<768, 256, 0, stream>>>(Wh, WhT);

    // xp = x @ Wx + bx  (whole-GPU bf16 WMMA GEMM, ~14us at 23.3 TB/s)
    xp_gemm_kernel<<<TBROWS / 64, 256, 0, stream>>>(x, WxT, bx, xp);

    // Sequential GRU scan: one persistent WG; Wh hot in L2 (read once/step),
    // xp streamed by the Tensor Data Mover one step ahead into LDS.
    gru_scan_kernel<<<1, 512, 0, stream>>>(xp, WhT, bh, h0,
                                           out, out + B_SZ * H_SZ);
}
